// GraphTransformerLayer_20761871908966
// MI455X (gfx1250) — matrix operand order repaired and checked by reading, not compiled, not run on hardware
//
#include <hip/hip_runtime.h>
#include <math.h>

// ---------------- problem constants ----------------
#define NN   50000
#define NPAD 50016                 // NN rounded up to 32 (macro-tile M)
#define EE   800000
#define HID  256
#define HEADS 8
#define CDIM 32
#define EDIM 64
#define FF   1024

// ---------------- WMMA vector types ----------------
typedef __bf16 v16bf __attribute__((ext_vector_type(16)));
typedef __bf16 v8bf  __attribute__((ext_vector_type(8)));
typedef float  v8f   __attribute__((ext_vector_type(8)));

union ABu { v16bf v; uint4 u[2]; };

// =====================================================================
// fp32 -> bf16 cast
// =====================================================================
__global__ void cvt_kernel(const float* __restrict__ in, __bf16* __restrict__ out, long n) {
    long i = (long)blockIdx.x * blockDim.x + threadIdx.x;
    if (i < n) out[i] = (__bf16)in[i];
}

// =====================================================================
// WMMA bf16 GEMM, 2x2 register-blocked: each wave computes a 32x32 tile
// of C[M,Nout] = A[M,K] @ B[Nout,K]^T (+bias) (optional exact GELU).
//   - A rows must be readable up to Mtiles32*32 (padded workspace).
//   - Stores are row-guarded against M.
// Fragment layouts per CDNA5 ISA 7.12.2 (wave32):
//   A 16x32 bf16 : m = lane&15 ; halves 0..7 -> k = kb+(lane>=16?8:0)+h, halves 8..15 -> +16
//   B 32x16 bf16 : n = lane&15 ; halves 0..7 -> k = kb+(lane>=16?8:0)+h, halves 8..15 -> +16
//   C/D 16x16 f32: col = lane&15 ; VGPR r -> row = r + (lane>=16?8:0)
// =====================================================================
__device__ __forceinline__ v8f wmma_bf16(const ABu& a, const ABu& b, v8f c) {
    return __builtin_amdgcn_wmma_f32_16x16x32_bf16(
               false, a.v, false, b.v, (short)0, c, false, false);
}

__device__ __forceinline__ void store_tile(v8f acc, int rowbase, int col, int grp,
                                           int M, int Nout, const float* bias, int gelu,
                                           float* Cf, __bf16* Cb) {
    const float bv = bias ? bias[col] : 0.0f;
    #pragma unroll
    for (int r = 0; r < 8; ++r) {
        const int row = rowbase + r + grp * 8;
        if (row < M) {
            float v = acc[r] + bv;
            if (gelu) v = 0.5f * v * (1.0f + erff(v * 0.70710678118654752f));
            if (Cf) Cf[(size_t)row * Nout + col] = v;
            if (Cb) Cb[(size_t)row * Nout + col] = (__bf16)v;
        }
    }
}

__global__ void wmma_gemm_bf16(const __bf16* __restrict__ A,
                               const __bf16* __restrict__ B,
                               const float* __restrict__ bias,
                               float* __restrict__ Cf,
                               __bf16* __restrict__ Cb,
                               int M, int Mtiles32, int K, int Nout, int gelu) {
    const int lane = threadIdx.x;                       // 0..31
    const int tn   = blockIdx.x;                        // 32-col macro tile
    const int tm   = blockIdx.y * blockDim.y + threadIdx.y;
    if (tm >= Mtiles32) return;                         // wave-uniform exit

    const int mr  = lane & 15;
    const int grp = lane >> 4;                          // lane half-group

    const __bf16* a0 = A + (size_t)(tm * 32 + mr) * K + grp * 8;
    const __bf16* a1 = a0 + (size_t)16 * K;
    const __bf16* b0 = B + (size_t)(tn * 32 + mr) * K + grp * 8;
    const __bf16* b1 = b0 + (size_t)16 * K;

    v8f acc00 = {}, acc01 = {}, acc10 = {}, acc11 = {};
    for (int kb = 0; kb < K; kb += 32) {
        ABu fa0, fa1, fb0, fb1;
        fa0.u[0] = *(const uint4*)(a0);  fa0.u[1] = *(const uint4*)(a0 + 16);
        fa1.u[0] = *(const uint4*)(a1);  fa1.u[1] = *(const uint4*)(a1 + 16);
        fb0.u[0] = *(const uint4*)(b0);  fb0.u[1] = *(const uint4*)(b0 + 16);
        fb1.u[0] = *(const uint4*)(b1);  fb1.u[1] = *(const uint4*)(b1 + 16);
        a0 += 32; a1 += 32; b0 += 32; b1 += 32;
        __builtin_prefetch(a0, 0, 1);                   // global_prefetch_b8
        __builtin_prefetch(b0, 0, 1);
        acc00 = wmma_bf16(fa0, fb0, acc00);
        acc01 = wmma_bf16(fa0, fb1, acc01);
        acc10 = wmma_bf16(fa1, fb0, acc10);
        acc11 = wmma_bf16(fa1, fb1, acc11);
    }

    const int r0 = tm * 32, r1 = tm * 32 + 16;
    const int c0 = tn * 32 + mr, c1 = tn * 32 + 16 + mr;
    store_tile(acc00, r0, c0, grp, M, Nout, bias, gelu, Cf, Cb);
    store_tile(acc01, r0, c1, grp, M, Nout, bias, gelu, Cf, Cb);
    store_tile(acc10, r1, c0, grp, M, Nout, bias, gelu, Cf, Cb);
    store_tile(acc11, r1, c1, grp, M, Nout, bias, gelu, Cf, Cb);
}

// =====================================================================
// init: amax = -inf, denom = 0, out-accum = 0
// =====================================================================
__global__ void init_kernel(float* amax, float* denom, float* outacc) {
    long i = (long)blockIdx.x * blockDim.x + threadIdx.x;
    if (i < (long)NN * HID) outacc[i] = 0.0f;
    if (i < (long)NN * HEADS) { amax[i] = -__builtin_inff(); denom[i] = 0.0f; }
}

// =====================================================================
// float atomic max via CAS (handles -inf init and negatives)
// =====================================================================
__device__ __forceinline__ void atomicMaxF(float* addr, float val) {
    int old = __float_as_int(*addr);
    while (__int_as_float(old) < val) {
        int assumed = old;
        old = atomicCAS((int*)addr, assumed, __float_as_int(val));
        if (old == assumed) break;
    }
}

// =====================================================================
// pass 1: alpha[e,h] = <q[dst,h,:], k[src,h,:] + e[e,h,:]> / sqrt(C)
// =====================================================================
__global__ void edge_logits_kernel(const int* __restrict__ ei,
                                   const float* __restrict__ q,
                                   const float* __restrict__ k,
                                   const __bf16* __restrict__ eb,
                                   float* __restrict__ alpha,
                                   float* __restrict__ amax) {
    long idx = (long)blockIdx.x * blockDim.x + threadIdx.x;
    if (idx >= (long)EE * HEADS) return;
    const int eid = (int)(idx >> 3);
    const int h   = (int)(idx & 7);
    const int src = ei[eid];
    const int dst = ei[EE + eid];

    const float* qd = q + (size_t)dst * HID + h * CDIM;
    const float* ks = k + (size_t)src * HID + h * CDIM;
    const v8bf*  ev = (const v8bf*)(eb + (size_t)eid * HID + h * CDIM);  // 16B aligned

    float acc = 0.0f;
    #pragma unroll
    for (int c8 = 0; c8 < CDIM / 8; ++c8) {
        const v8bf e8 = ev[c8];
        const float4 q0 = ((const float4*)qd)[c8 * 2 + 0];
        const float4 q1 = ((const float4*)qd)[c8 * 2 + 1];
        const float4 k0 = ((const float4*)ks)[c8 * 2 + 0];
        const float4 k1 = ((const float4*)ks)[c8 * 2 + 1];
        acc += q0.x * (k0.x + (float)e8[0]) + q0.y * (k0.y + (float)e8[1])
             + q0.z * (k0.z + (float)e8[2]) + q0.w * (k0.w + (float)e8[3])
             + q1.x * (k1.x + (float)e8[4]) + q1.y * (k1.y + (float)e8[5])
             + q1.z * (k1.z + (float)e8[6]) + q1.w * (k1.w + (float)e8[7]);
    }
    const float a = acc * 0.17677669529663689f;   // 1/sqrt(32)
    alpha[idx] = a;
    atomicMaxF(&amax[(size_t)dst * HEADS + h], a);
}

// =====================================================================
// pass 2: alpha <- exp(alpha - amax[dst]);  denom[dst] += alpha
// =====================================================================
__global__ void edge_exp_kernel(const int* __restrict__ ei,
                                float* __restrict__ alpha,
                                const float* __restrict__ amax,
                                float* __restrict__ denom) {
    long idx = (long)blockIdx.x * blockDim.x + threadIdx.x;
    if (idx >= (long)EE * HEADS) return;
    const int eid = (int)(idx >> 3);
    const int h   = (int)(idx & 7);
    const int dst = ei[EE + eid];
    const float ex = __expf(alpha[idx] - amax[(size_t)dst * HEADS + h]);
    alpha[idx] = ex;
    atomicAdd(&denom[(size_t)dst * HEADS + h], ex);
}

// =====================================================================
// pass 3: out[dst,c] += (alpha/denom) * (v[src,c] + e[eid,c])
//         one thread per (edge, channel) -> 1 coalesced atomic each
// =====================================================================
__global__ void edge_scatter_kernel(const int* __restrict__ ei,
                                    const float* __restrict__ v,
                                    const __bf16* __restrict__ eb,
                                    const float* __restrict__ alpha,
                                    const float* __restrict__ denom,
                                    float* __restrict__ outacc) {
    long idx = (long)blockIdx.x * blockDim.x + threadIdx.x;
    if (idx >= (long)EE * HID) return;
    const int eid = (int)(idx >> 8);
    const int c   = (int)(idx & 255);
    const int h   = c >> 5;
    const int src = ei[eid];
    const int dst = ei[EE + eid];
    const float w = alpha[(size_t)eid * HEADS + h]
                  / (denom[(size_t)dst * HEADS + h] + 1e-16f);
    const float val = w * (v[(size_t)src * HID + c] + (float)eb[(size_t)eid * HID + c]);
    atomicAdd(&outacc[(size_t)dst * HID + c], val);
}

// =====================================================================
// block-wide sum over 256 threads
// =====================================================================
__device__ __forceinline__ float block_sum(float val, float* sm) {
    const int t = threadIdx.x;
    sm[t] = val;
    __syncthreads();
    for (int s = 128; s > 0; s >>= 1) {
        if (t < s) sm[t] += sm[t + s];
        __syncthreads();
    }
    const float r = sm[0];
    __syncthreads();
    return r;
}

// =====================================================================
// beta gate + residual + LayerNorm1 (one block of 256 threads per node)
// =====================================================================
__global__ void beta_ln1_kernel(const float* __restrict__ x,
                                const float* __restrict__ outacc,
                                const float* __restrict__ xr,
                                const float* __restrict__ Wbeta,
                                const float* __restrict__ g,
                                const float* __restrict__ bpar,
                                float* __restrict__ yf,
                                __bf16* __restrict__ yb) {
    __shared__ float sm[HID];
    const int nid = blockIdx.x;
    const int c   = threadIdx.x;
    const size_t base = (size_t)nid * HID;

    const float o = outacc[base + c];
    const float r = xr[base + c];
    const float part = o * Wbeta[c] + r * Wbeta[HID + c] + (o - r) * Wbeta[2 * HID + c];
    const float s    = block_sum(part, sm);
    const float beta = 1.0f / (1.0f + __expf(-s));
    const float h    = beta * r + (1.0f - beta) * o;
    const float y    = x[base + c] + h;

    const float mu  = block_sum(y, sm) * (1.0f / HID);
    const float d   = y - mu;
    const float var = block_sum(d * d, sm) * (1.0f / HID);
    const float yn  = d * rsqrtf(var + 1e-5f) * g[c] + bpar[c];

    yf[base + c] = yn;
    yb[base + c] = (__bf16)yn;
}

// =====================================================================
// residual + LayerNorm2 -> final output
// =====================================================================
__global__ void ln2_kernel(const float* __restrict__ xln1,
                           const float* __restrict__ h2,
                           const float* __restrict__ g,
                           const float* __restrict__ bpar,
                           float* __restrict__ out) {
    __shared__ float sm[HID];
    const int nid = blockIdx.x;
    const int c   = threadIdx.x;
    const size_t base = (size_t)nid * HID;

    const float y   = xln1[base + c] + h2[base + c];
    const float mu  = block_sum(y, sm) * (1.0f / HID);
    const float d   = y - mu;
    const float var = block_sum(d * d, sm) * (1.0f / HID);
    out[base + c] = d * rsqrtf(var + 1e-5f) * g[c] + bpar[c];
}

// =====================================================================
// host launch
// =====================================================================
static inline void* ws_alloc(char* base, size_t& off, size_t bytes) {
    void* p = base + off;
    off += (bytes + 255) & ~(size_t)255;
    return p;
}

extern "C" void kernel_launch(void* const* d_in, const int* in_sizes, int n_in,
                              void* d_out, int out_size, void* d_ws, size_t ws_size,
                              hipStream_t stream) {
    const float* x      = (const float*)d_in[0];
    const int*   ei     = (const int*)d_in[1];
    const float* eattr  = (const float*)d_in[2];
    const float* Wq     = (const float*)d_in[3];
    const float* bq     = (const float*)d_in[4];
    const float* Wk     = (const float*)d_in[5];
    const float* bk     = (const float*)d_in[6];
    const float* Wv     = (const float*)d_in[7];
    const float* bv     = (const float*)d_in[8];
    const float* We     = (const float*)d_in[9];
    const float* Wskip  = (const float*)d_in[10];
    const float* bskip  = (const float*)d_in[11];
    const float* Wbeta  = (const float*)d_in[12];
    const float* ln1g   = (const float*)d_in[13];
    const float* ln1b   = (const float*)d_in[14];
    const float* ln2g   = (const float*)d_in[15];
    const float* ln2b   = (const float*)d_in[16];
    const float* W1     = (const float*)d_in[17];
    const float* b1     = (const float*)d_in[18];
    const float* W2     = (const float*)d_in[19];
    const float* b2     = (const float*)d_in[20];
    float* out = (float*)d_out;

    // -------- workspace layout (A-side bf16 buffers padded to 32-row tiles) --------
    char*  base = (char*)d_ws;
    size_t off  = 0;
    __bf16* xb    = (__bf16*)ws_alloc(base, off, (size_t)NPAD * HID * 2);
    __bf16* WqB   = (__bf16*)ws_alloc(base, off, (size_t)HID * HID * 2);
    __bf16* WkB   = (__bf16*)ws_alloc(base, off, (size_t)HID * HID * 2);
    __bf16* WvB   = (__bf16*)ws_alloc(base, off, (size_t)HID * HID * 2);
    __bf16* WsB   = (__bf16*)ws_alloc(base, off, (size_t)HID * HID * 2);
    __bf16* WeB   = (__bf16*)ws_alloc(base, off, (size_t)HID * EDIM * 2);
    __bf16* W1B   = (__bf16*)ws_alloc(base, off, (size_t)FF * HID * 2);
    __bf16* W2B   = (__bf16*)ws_alloc(base, off, (size_t)HID * FF * 2);
    __bf16* eab   = (__bf16*)ws_alloc(base, off, (size_t)EE * EDIM * 2);
    float*  qf    = (float*) ws_alloc(base, off, (size_t)NN * HID * 4);
    float*  kf    = (float*) ws_alloc(base, off, (size_t)NN * HID * 4);
    float*  vf    = (float*) ws_alloc(base, off, (size_t)NN * HID * 4);
    float*  xrf   = (float*) ws_alloc(base, off, (size_t)NN * HID * 4);
    __bf16* eb    = (__bf16*)ws_alloc(base, off, (size_t)EE * HID * 2);
    float*  alpha = (float*) ws_alloc(base, off, (size_t)EE * HEADS * 4);
    float*  amax  = (float*) ws_alloc(base, off, (size_t)NN * HEADS * 4);
    float*  denom = (float*) ws_alloc(base, off, (size_t)NN * HEADS * 4);
    float*  oacc  = (float*) ws_alloc(base, off, (size_t)NN * HID * 4);
    float*  ln1f  = (float*) ws_alloc(base, off, (size_t)NN * HID * 4);
    __bf16* ln1bf = (__bf16*)ws_alloc(base, off, (size_t)NPAD * HID * 2);
    __bf16* h1b   = (__bf16*)ws_alloc(base, off, (size_t)NPAD * FF * 2);
    float*  h2f   = (float*) ws_alloc(base, off, (size_t)NN * HID * 4);
    (void)ws_size; (void)in_sizes; (void)n_in; (void)out_size;

    auto cvt = [&](const float* src, __bf16* dst, long n) {
        cvt_kernel<<<(unsigned)((n + 255) / 256), 256, 0, stream>>>(src, dst, n);
    };

    // -------- fp32 -> bf16 casts --------
    cvt(x,     xb,  (long)NN * HID);
    cvt(Wq,    WqB, (long)HID * HID);
    cvt(Wk,    WkB, (long)HID * HID);
    cvt(Wv,    WvB, (long)HID * HID);
    cvt(Wskip, WsB, (long)HID * HID);
    cvt(We,    WeB, (long)HID * EDIM);
    cvt(W1,    W1B, (long)FF * HID);
    cvt(W2,    W2B, (long)HID * FF);
    cvt(eattr, eab, (long)EE * EDIM);

    // -------- projection GEMMs (WMMA bf16, 32x32 per wave) --------
    const dim3 gb(32, 4);
    const int MtN = NPAD / 32;                       // 1563 macro tiles over nodes
    {   // q, k, v, skip : [N,256] = xb @ W[256,256]^T
        const dim3 gg(HID / 32, (MtN + 3) / 4);
        wmma_gemm_bf16<<<gg, gb, 0, stream>>>(xb, WqB, bq,    qf,  nullptr, NN, MtN, HID, HID, 0);
        wmma_gemm_bf16<<<gg, gb, 0, stream>>>(xb, WkB, bk,    kf,  nullptr, NN, MtN, HID, HID, 0);
        wmma_gemm_bf16<<<gg, gb, 0, stream>>>(xb, WvB, bv,    vf,  nullptr, NN, MtN, HID, HID, 0);
        wmma_gemm_bf16<<<gg, gb, 0, stream>>>(xb, WsB, bskip, xrf, nullptr, NN, MtN, HID, HID, 0);
    }
    {   // e : [E,256] = edge_attr[E,64] @ We[256,64]^T  (no bias, bf16 out)
        const int MtE = EE / 32;
        const dim3 gg(HID / 32, (MtE + 3) / 4);
        wmma_gemm_bf16<<<gg, gb, 0, stream>>>(eab, WeB, nullptr, nullptr, eb, EE, MtE, EDIM, HID, 0);
    }

    // -------- segment softmax over incoming edges --------
    init_kernel<<<(NN * HID + 255) / 256, 256, 0, stream>>>(amax, denom, oacc);

    const long eh = (long)EE * HEADS;
    edge_logits_kernel<<<(unsigned)((eh + 255) / 256), 256, 0, stream>>>(ei, qf, kf, eb, alpha, amax);
    edge_exp_kernel<<<(unsigned)((eh + 255) / 256), 256, 0, stream>>>(ei, alpha, amax, denom);
    const long ec = (long)EE * HID;
    edge_scatter_kernel<<<(unsigned)((ec + 255) / 256), 256, 0, stream>>>(ei, vf, eb, alpha, denom, oacc);

    // -------- beta gate + residual + LN1 --------
    beta_ln1_kernel<<<NN, HID, 0, stream>>>(x, oacc, xrf, Wbeta, ln1g, ln1b, ln1f, ln1bf);

    // -------- FFN (WMMA bf16, exact GELU fused into GEMM1) --------
    {   // h1 = gelu(ln1 @ W1^T + b1) : [N,1024] bf16, feeds GEMM2 directly
        const dim3 gg(FF / 32, (MtN + 3) / 4);
        wmma_gemm_bf16<<<gg, gb, 0, stream>>>(ln1bf, W1B, b1, nullptr, h1b, NN, MtN, HID, FF, 1);
    }
    {   // h2 = h1 @ W2^T + b2 : [N,256] fp32
        const dim3 gg(HID / 32, (MtN + 3) / 4);
        wmma_gemm_bf16<<<gg, gb, 0, stream>>>(h1b, W2B, b2, h2f, nullptr, NN, MtN, FF, HID, 0);
    }

    // -------- residual + LN2 -> output --------
    ln2_kernel<<<NN, HID, 0, stream>>>(ln1f, h2f, ln2g, ln2b, out);
}
